// DeltaNet_31877247271438
// MI455X (gfx1250) — matrix operand order repaired and checked by reading, not compiled, not run on hardware
//
#include <hip/hip_runtime.h>
#include <math.h>

typedef __bf16 bf16;
typedef __attribute__((ext_vector_type(16))) __bf16 v16bf;
typedef __attribute__((ext_vector_type(8)))  __bf16 v8bf;
typedef __attribute__((ext_vector_type(8)))  float  v8f;

#define DEV static __device__ __forceinline__

DEV float b2f(bf16 x) { return (float)x; }
DEV bf16  f2b(float x) { return (bf16)x; }

DEV v8f wmma_bf16(v16bf a, v16bf b, v8f c) {
    return __builtin_amdgcn_wmma_f32_16x16x32_bf16(
        /*neg_a=*/false, a, /*neg_b=*/false, b,
        /*c_mod=*/(short)0, c, /*reuse_a=*/false, /*reuse_b=*/false);
}

// ---------------- WMMA fragment helpers (wave32 layouts, ISA 7.12.2) ----------------
// A fragment: 16x32 (MxK), row-major source, leading dim lda (elements).
// lane m = lane&15; k = 16*(j>=8) + 8*(lane>>4) + (j&7) -> two contiguous 16B runs.
DEV v16bf load_frag_a(const bf16* p, int lda) {
    int lane = threadIdx.x & 31;
    int m    = lane & 15;
    int kb   = ((lane >> 4) & 1) * 8;
    const bf16* p0 = p + (size_t)m * lda + kb;
    v8bf lo = *(const v8bf*)(p0);
    v8bf hi = *(const v8bf*)(p0 + 16);
    v16bf r;
#pragma unroll
    for (int j = 0; j < 8; ++j) { r[j] = lo[j]; r[j + 8] = hi[j]; }
    return r;
}

// B fragment from N-major storage (B^T row-major, ld elems per N-row).
// lane n = lane&15; k = 16*(j>=8) + 8*(lane>>4) + (j&7) -> two contiguous 16B runs.
DEV v16bf load_frag_bT(const bf16* p, int ld) {
    int lane = threadIdx.x & 31;
    int n    = lane & 15;
    int k0   = ((lane >> 4) & 1) * 8;
    const bf16* q = p + (size_t)n * ld + k0;
    v8bf lo = *(const v8bf*)(q);
    v8bf hi = *(const v8bf*)(q + 16);
    v16bf r;
#pragma unroll
    for (int j = 0; j < 8; ++j) { r[j] = lo[j]; r[j + 8] = hi[j]; }
    return r;
}

// C/D fragment: 16x16 fp32. lane n = lane&15; row m = 8*(lane>>4) + r.
DEV v8f load_frag_c(const float* p, int ldc) {
    int lane = threadIdx.x & 31;
    int n    = lane & 15;
    int mb   = ((lane >> 4) & 1) * 8;
    v8f r;
#pragma unroll
    for (int j = 0; j < 8; ++j) r[j] = p[(size_t)(mb + j) * ldc + n];
    return r;
}

DEV void store_frag_c(float* p, int ldc, v8f a) {
    int lane = threadIdx.x & 31;
    int n    = lane & 15;
    int mb   = ((lane >> 4) & 1) * 8;
#pragma unroll
    for (int r = 0; r < 8; ++r) p[(size_t)(mb + r) * ldc + n] = a[r];
}

// store C fragment transposed as bf16 (row = n, contiguous in m): ONE 16B store per lane.
DEV void store_frag_cT_bf16(bf16* p, int ld, v8f a) {
    int lane = threadIdx.x & 31;
    int n    = lane & 15;
    int mb   = ((lane >> 4) & 1) * 8;
    v8bf v;
#pragma unroll
    for (int r = 0; r < 8; ++r) v[r] = f2b(a[r]);
    *(v8bf*)(p + (size_t)n * ld + mb) = v;
}

// ---------------- Generic tiled bf16 WMMA GEMM: C[M,N] = A[M,K]*B[K,N] ----------------
// Requires M%128==0, N%64==0, K%32==0. Block = 256 threads (8 waves), tile 128x64x32.
// B tile staged TRANSPOSED in LDS so B-fragments are contiguous b128 loads.
__global__ __launch_bounds__(256) void gemm_bf16_kernel(
    const bf16* __restrict__ A, const bf16* __restrict__ B, float* __restrict__ C,
    int M, int N, int K)
{
    __shared__ __align__(16) bf16 As[128 * 32];
    __shared__ __align__(16) bf16 BsT[64 * 32];
    int tid  = threadIdx.x;
    int wave = tid >> 5;
    int wm = wave & 3, wn = wave >> 2;          // 4x2 waves, each 32x32 output
    int m0 = blockIdx.y * 128, n0 = blockIdx.x * 64;
    int arow = tid >> 1, acol = (tid & 1) * 16; // 128 rows x 32 cols, 32B per thread
    int brow = tid >> 3, bcol = (tid & 7) * 8;  // 32 k-rows x 64 n-cols
    v8f acc00 = {}; v8f acc01 = {}; v8f acc10 = {}; v8f acc11 = {};
    for (int k0 = 0; k0 < K; k0 += 32) {
        const bf16* ap = &A[(size_t)(m0 + arow) * K + k0 + acol];
        *(int4*)&As[arow * 32 + acol]     = *(const int4*)(ap);
        *(int4*)&As[arow * 32 + acol + 8] = *(const int4*)(ap + 8);
        v8bf bv = *(const v8bf*)&B[(size_t)(k0 + brow) * N + n0 + bcol];
#pragma unroll
        for (int e = 0; e < 8; ++e) BsT[(bcol + e) * 32 + brow] = bv[e];
        if (k0 + 32 < K) {
            __builtin_prefetch(&A[(size_t)(m0 + arow) * K + k0 + 32 + acol], 0, 1);
            __builtin_prefetch(&B[(size_t)(k0 + 32 + brow) * N + n0 + bcol], 0, 1);
        }
        __syncthreads();
        v16bf a0 = load_frag_a(&As[(wm * 32) * 32], 32);
        v16bf a1 = load_frag_a(&As[(wm * 32 + 16) * 32], 32);
        v16bf b0 = load_frag_bT(&BsT[(wn * 32) * 32], 32);
        v16bf b1 = load_frag_bT(&BsT[(wn * 32 + 16) * 32], 32);
        acc00 = wmma_bf16(a0, b0, acc00);
        acc01 = wmma_bf16(a0, b1, acc01);
        acc10 = wmma_bf16(a1, b0, acc10);
        acc11 = wmma_bf16(a1, b1, acc11);
        __syncthreads();
    }
    float* cp = &C[(size_t)(m0 + wm * 32) * N + n0 + wn * 32];
    store_frag_c(cp, N, acc00);
    store_frag_c(cp + 16, N, acc01);
    store_frag_c(cp + (size_t)16 * N, N, acc10);
    store_frag_c(cp + (size_t)16 * N + 16, N, acc11);
}

// ---------------- elementwise / small kernels ----------------
__global__ void cvt_f32_bf16_kernel(const float* __restrict__ x, bf16* __restrict__ y, size_t n) {
    size_t i = (size_t)blockIdx.x * blockDim.x + threadIdx.x;
    if (i < n) y[i] = f2b(x[i]);
}

// causal depthwise conv k=4 over L, channels=1024, + SiLU.
__global__ void conv4_silu_kernel(const float* __restrict__ x, const float* __restrict__ w,
                                  float* __restrict__ y, size_t n) {
    size_t i = (size_t)blockIdx.x * blockDim.x + threadIdx.x;
    if (i >= n) return;
    int c = (int)(i & 1023);
    int l = (int)((i >> 10) & 2047);
    size_t bl = i >> 10;  // b*2048 + l
    float s = 0.f;
#pragma unroll
    for (int t = 0; t < 4; ++t) {
        int l0 = l - 3 + t;
        if (l0 >= 0) s += x[((bl - l + l0) << 10) + c] * w[c * 4 + t];
    }
    y[i] = s / (1.f + expf(-s));
}

__global__ void beta_kernel(const float* __restrict__ hs, const float* __restrict__ Wb,
                            float* __restrict__ beta, int n) {
    int i = blockIdx.x * blockDim.x + threadIdx.x;
    if (i >= n) return;
    int row = i >> 2, h = i & 3;
    const float* x = hs + (size_t)row * 1024;
    float s = 0.f;
    for (int k = 0; k < 1024; ++k) s += x[k] * Wb[k * 4 + h];
    beta[i] = 1.f / (1.f + expf(-s));
}

// l2norm q/k per (b,l,h), beta-scale; emit bf16 chunk layouts:
//  qn,kn,kb: [chunk r][d] row-major ; kbT,vbT,kT: [d][chunk r] (d-major, B/A-transposed uses)
// grid = B*L (8192), block 128 = 4 waves = 4 heads
__global__ __launch_bounds__(128) void delta_prep_kernel(
    const float* __restrict__ qa, const float* __restrict__ ka, const float* __restrict__ va,
    const float* __restrict__ beta,
    bf16* __restrict__ qn, bf16* __restrict__ kn, bf16* __restrict__ kb,
    bf16* __restrict__ kbT, bf16* __restrict__ vbT, bf16* __restrict__ kT)
{
    int row  = blockIdx.x;          // b*2048 + l
    int h    = threadIdx.x >> 5;
    int lane = threadIdx.x & 31;
    int b = row >> 11, l = row & 2047;
    int nck = l >> 5, r = l & 31;
    int bh = b * 4 + h;
    size_t src = (size_t)row * 1024 + h * 256;
    float qv[8], kv[8];
    float sq = 0.f, sk = 0.f;
#pragma unroll
    for (int e = 0; e < 8; ++e) {
        int d = lane + e * 32;
        qv[e] = qa[src + d]; kv[e] = ka[src + d];
        sq += qv[e] * qv[e]; sk += kv[e] * kv[e];
    }
#pragma unroll
    for (int o = 16; o > 0; o >>= 1) { sq += __shfl_xor(sq, o, 32); sk += __shfl_xor(sk, o, 32); }
    float rq = rsqrtf(sq + 1e-6f), rk = rsqrtf(sk + 1e-6f);
    float bet = beta[(size_t)row * 4 + h];
    size_t cb  = ((size_t)(bh * 64 + nck) * 32 + r) * 256;
    size_t tb0 = ((size_t)(bh * 64 + nck)) * 256 * 32;
#pragma unroll
    for (int e = 0; e < 8; ++e) {
        int d = lane + e * 32;
        float qnv = qv[e] * rq, knv = kv[e] * rk;
        qn[cb + d] = f2b(qnv);
        kn[cb + d] = f2b(knv);
        kb[cb + d] = f2b(knv * bet);
        kbT[tb0 + (size_t)d * 32 + r] = f2b(knv * bet);
        vbT[tb0 + (size_t)d * 32 + r] = f2b(va[src + d] * bet);
        kT [tb0 + (size_t)d * 32 + r] = f2b(knv);
    }
}

// Per-chunk: a_loc = (q k^T)*incl_lower ; T = (I + tril(beta k k^T,-1))^{-1} via fwd subst;
// u = T @ (v*beta) ; wneg = -(T @ (k*beta)).   grid = 16*64 chunks, block 128 = 4 waves
__global__ __launch_bounds__(128) void delta_chunk_kernel(
    const bf16* __restrict__ qn, const bf16* __restrict__ kn, const bf16* __restrict__ kb,
    const bf16* __restrict__ kbT, const bf16* __restrict__ vbT,
    float* __restrict__ u, bf16* __restrict__ wneg, bf16* __restrict__ aloc)
{
    __shared__ __align__(16) float Tm[1024];
    __shared__ __align__(16) bf16  Tb[1024];
    int tid = threadIdx.x, lane = tid & 31, wv = tid >> 5;
    int cid = blockIdx.x;
    size_t cb = (size_t)cid * 32 * 256;
    size_t ab = (size_t)cid * 1024;
    int mt = wv & 1, nt = wv >> 1;
    {   // a_loc (inclusive lower mask); B = k^T is just kn in its N-major chunk layout
        v8f acc = {};
#pragma unroll
        for (int kk = 0; kk < 8; ++kk) {
            v16bf a  = load_frag_a (qn + cb + (size_t)(mt * 16) * 256 + kk * 32, 256);
            v16bf bb = load_frag_bT(kn + cb + (size_t)(nt * 16) * 256 + kk * 32, 256);
            acc = wmma_bf16(a, bb, acc);
        }
        int nn = nt * 16 + (lane & 15);
        int mb = mt * 16 + ((lane >> 4) & 1) * 8;
#pragma unroll
        for (int r = 0; r < 8; ++r) {
            int m = mb + r;
            aloc[ab + m * 32 + nn] = f2b(m >= nn ? acc[r] : 0.f);
        }
    }
    {   // T0 = -(kb @ k^T) * strict_lower
        v8f acc = {};
#pragma unroll
        for (int kk = 0; kk < 8; ++kk) {
            v16bf a  = load_frag_a (kb + cb + (size_t)(mt * 16) * 256 + kk * 32, 256);
            v16bf bb = load_frag_bT(kn + cb + (size_t)(nt * 16) * 256 + kk * 32, 256);
            acc = wmma_bf16(a, bb, acc);
        }
        int nn = nt * 16 + (lane & 15);
        int mb = mt * 16 + ((lane >> 4) & 1) * 8;
#pragma unroll
        for (int r = 0; r < 8; ++r) {
            int m = mb + r;
            Tm[m * 32 + nn] = (m > nn) ? -acc[r] : 0.f;
        }
    }
    __syncthreads();
    // forward substitution: row i += row_i-partial products of already-final rows
    for (int i = 1; i < 32; ++i) {
        float s = 0.f;
        if (tid < i) for (int t = tid + 1; t < i; ++t) s += Tm[i * 32 + t] * Tm[t * 32 + tid];
        __syncthreads();
        if (tid < i) Tm[i * 32 + tid] += s;
        __syncthreads();
    }
    for (int e = tid; e < 1024; e += 128) {
        float v = Tm[e];
        if ((e >> 5) == (e & 31)) v += 1.f;     // + identity
        Tb[e] = f2b(v);
    }
    __syncthreads();
    // u = T @ vb ; wneg = -(T @ kb)   (B operands from d-major copies -> contiguous frags)
    for (int t = wv; t < 32; t += 4) {
        int m2 = t & 1, n2 = t >> 1;
        v8f z = {};
        v16bf a  = load_frag_a(&Tb[(m2 * 16) * 32], 32);
        v16bf bv = load_frag_bT(vbT + cb + (size_t)(n2 * 16) * 32, 32);
        v8f au = wmma_bf16(a, bv, z);
        store_frag_c(u + cb + (size_t)(m2 * 16) * 256 + n2 * 16, 256, au);
        v16bf bk = load_frag_bT(kbT + cb + (size_t)(n2 * 16) * 32, 32);
        v8f aw = wmma_bf16(a, bk, z);
        int nn = lane & 15, mb = ((lane >> 4) & 1) * 8;
        bf16* wp = wneg + cb + (size_t)(m2 * 16) * 256 + n2 * 16;
#pragma unroll
        for (int r = 0; r < 8; ++r) wp[(size_t)(mb + r) * 256 + nn] = f2b(-aw[r]);
    }
}

// Sequential chunk scan. State S (256 x 128 slice) in LDS: fp32 + TRANSPOSED bf16 shadow
// SbT[dv=128][dk=256]; UbT[dv=128][r=32]. grid = 16 heads * 2 dv-halves, block 256 (8 waves).
__global__ __launch_bounds__(256) void delta_scan_kernel(
    const bf16* __restrict__ qn, const bf16* __restrict__ wneg, const bf16* __restrict__ aloc,
    const bf16* __restrict__ kT, const float* __restrict__ u, float* __restrict__ dout)
{
    extern __shared__ char smem[];
    float* S   = (float*)smem;                                  // 256*128*4 = 131072
    bf16*  SbT = (bf16*)(smem + 131072);                        // 128*256*2 =  65536
    bf16*  UbT = (bf16*)(smem + 131072 + 65536);                // 128*32*2  =   8192
    int tid = threadIdx.x, wv = tid >> 5, lane = tid & 31;
    int bh  = blockIdx.x >> 1;
    int dv0 = (blockIdx.x & 1) * 128;
    int b = bh >> 2, h = bh & 3;
    for (int i = tid; i < 256 * 128; i += 256) { S[i] = 0.f; SbT[i] = f2b(0.f); }
    __syncthreads();
    for (int n = 0; n < 64; ++n) {
        size_t cb = ((size_t)(bh * 64 + n)) * 32 * 256;
        size_t ab = ((size_t)(bh * 64 + n)) * 1024;
        // step1: Ub = u - w @ S   (w stored pre-negated; bf16 WMMA has no A-negate)
        for (int t = wv; t < 16; t += 8) {
            int mt = t & 1, nt = t >> 1;
            v8f acc = load_frag_c(u + cb + (size_t)(mt * 16) * 256 + dv0 + nt * 16, 256);
#pragma unroll
            for (int kk = 0; kk < 8; ++kk) {
                v16bf a  = load_frag_a (wneg + cb + (size_t)(mt * 16) * 256 + kk * 32, 256);
                v16bf bb = load_frag_bT(SbT + (size_t)(nt * 16) * 256 + kk * 32, 256);
                acc = wmma_bf16(a, bb, acc);
            }
            store_frag_cT_bf16(UbT + (size_t)(nt * 16) * 32 + mt * 16, 32, acc);
        }
        __syncthreads();
        // step2: o = q @ S + a_loc @ Ub  -> delta_out (b,l,h,dv)
        for (int t = wv; t < 16; t += 8) {
            int mt = t & 1, nt = t >> 1;
            v8f acc = {};
#pragma unroll
            for (int kk = 0; kk < 8; ++kk) {
                v16bf a  = load_frag_a (qn + cb + (size_t)(mt * 16) * 256 + kk * 32, 256);
                v16bf bb = load_frag_bT(SbT + (size_t)(nt * 16) * 256 + kk * 32, 256);
                acc = wmma_bf16(a, bb, acc);
            }
            v16bf a2 = load_frag_a (aloc + ab + (mt * 16) * 32, 32);
            v16bf b2 = load_frag_bT(UbT + (size_t)(nt * 16) * 32, 32);
            acc = wmma_bf16(a2, b2, acc);
            float* dp = dout + ((size_t)(b * 2048 + n * 32 + mt * 16)) * 1024 + h * 256 + dv0 + nt * 16;
            store_frag_c(dp, 1024, acc);
        }
        __syncthreads();
        // step3: S += k^T @ Ub  (refresh transposed bf16 shadow with single b128 stores)
        for (int t = wv * 16; t < wv * 16 + 16; ++t) {
            int mt = t >> 3, nt = t & 7;
            float* sp = S + (size_t)(mt * 16) * 128 + nt * 16;
            v8f acc = load_frag_c(sp, 128);
            v16bf a  = load_frag_a (kT + cb + (size_t)(mt * 16) * 32, 32);
            v16bf bb = load_frag_bT(UbT + (size_t)(nt * 16) * 32, 32);
            acc = wmma_bf16(a, bb, acc);
            int nn = lane & 15, mb = ((lane >> 4) & 1) * 8;
#pragma unroll
            for (int r = 0; r < 8; ++r) sp[(mb + r) * 128 + nn] = acc[r];
            store_frag_cT_bf16(SbT + (size_t)(nt * 16) * 256 + mt * 16, 256, acc);
        }
        __syncthreads();
    }
}

// multiscale causal convs, written in the torch-rearrange layout: row=(b,l,h), col=kd
__global__ void msconv_kernel(const float* __restrict__ v, const float* __restrict__ w3,
                              const float* __restrict__ w15, const float* __restrict__ w31,
                              bf16* __restrict__ msy, size_t n) {
    size_t i = (size_t)blockIdx.x * blockDim.x + threadIdx.x;
    if (i >= n) return;
    int cp = (int)(i % 3072);
    size_t bl = i / 3072;        // b*2048 + l
    int l = (int)(bl & 2047);
    int branch = cp >> 10;
    int c = cp & 1023;
    int taps; const float* wp;
    if (branch == 0) { taps = 3;  wp = w3;  }
    else if (branch == 1) { taps = 15; wp = w15; }
    else { taps = 31; wp = w31; }
    float s = 0.f;
    for (int t = 0; t < taps; ++t) {
        int l0 = l - (taps - 1) + t;
        if (l0 >= 0) s += v[((bl - l + l0) << 10) + c] * wp[c * taps + t];
    }
    int hh = cp / 768, kd = cp % 768;
    msy[((bl * 4 + hh)) * 768 + kd] = f2b(s);
}

// gate features: [hidden(1024) | mean|ms| (4) | mean|delta| (4) | mean|v| (4) | 20 zero-pad]
__global__ __launch_bounds__(256) void gate_kernel(
    const float* __restrict__ hs, const float* __restrict__ ms,
    const float* __restrict__ dl, const float* __restrict__ vd, bf16* __restrict__ gf)
{
    __shared__ float red[256];
    __shared__ float bn[12];
    int row = blockIdx.x, t = threadIdx.x;
    const float* srcs[3] = { ms + (size_t)row * 1024, dl + (size_t)row * 1024, vd + (size_t)row * 1024 };
    for (int s = 0; s < 3; ++s)
        for (int h = 0; h < 4; ++h) {
            red[t] = fabsf(srcs[s][h * 256 + t]);
            __syncthreads();
            for (int o = 128; o > 0; o >>= 1) { if (t < o) red[t] += red[t + o]; __syncthreads(); }
            if (t == 0) bn[s * 4 + h] = red[0] * (1.f / 256.f);
            __syncthreads();
        }
    bf16* g = gf + (size_t)row * 1056;
    for (int c = t; c < 1024; c += 256) g[c] = f2b(hs[(size_t)row * 1024 + c]);
    if (t < 32) g[1024 + t] = f2b(t < 12 ? bn[t] : 0.f);
}

__global__ void padw1_kernel(const float* __restrict__ w1, bf16* __restrict__ w1p, size_t n) {
    size_t i = (size_t)blockIdx.x * blockDim.x + threadIdx.x;
    if (i >= n) return;
    size_t r = i / 2048, c = i % 2048;
    w1p[i] = (r < 1036) ? f2b(w1[r * 2048 + c]) : f2b(0.f);
}

__global__ void biasgelu_kernel(const float* __restrict__ x, const float* __restrict__ b1,
                                bf16* __restrict__ y, size_t n) {
    size_t i = (size_t)blockIdx.x * blockDim.x + threadIdx.x;
    if (i >= n) return;
    float v = x[i] + b1[i % 2048];
    y[i] = f2b(0.5f * v * (1.f + erff(v * 0.70710678118f)));
}

// fusion_w2 + softmax gate + 3-way mix + RMSNorm -> bf16 rows for the final Wo GEMM
__global__ __launch_bounds__(256) void fuse_out_kernel(
    const bf16* __restrict__ hdn, const float* __restrict__ w2, const float* __restrict__ b2,
    const float* __restrict__ ms, const float* __restrict__ dl, const float* __restrict__ vd,
    const float* __restrict__ onorm, bf16* __restrict__ orms)
{
    __shared__ float red[256];
    __shared__ float lg[12];
    int row = blockIdx.x, t = threadIdx.x;
    const bf16* hr = hdn + (size_t)row * 2048;
    for (int j = 0; j < 12; ++j) {
        float p = 0.f;
        for (int k = t; k < 2048; k += 256) p += b2f(hr[k]) * w2[(size_t)k * 12 + j];
        red[t] = p;
        __syncthreads();
        for (int o = 128; o > 0; o >>= 1) { if (t < o) red[t] += red[t + o]; __syncthreads(); }
        if (t == 0) lg[j] = red[0] + b2[j];
        __syncthreads();
    }
    for (int h = 0; h < 4; ++h) {
        float l0 = lg[h * 3], l1 = lg[h * 3 + 1], l2 = lg[h * 3 + 2];
        float mx = fmaxf(l0, fmaxf(l1, l2));
        float e0 = expf(l0 - mx), e1 = expf(l1 - mx), e2 = expf(l2 - mx);
        float inv = 1.f / (e0 + e1 + e2);
        size_t ix = (size_t)row * 1024 + h * 256 + t;
        float o = (e0 * ms[ix] + e1 * dl[ix] + e2 * vd[ix]) * inv;
        red[t] = o * o;
        __syncthreads();
        for (int oo = 128; oo > 0; oo >>= 1) { if (t < oo) red[t] += red[t + oo]; __syncthreads(); }
        float scale = rsqrtf(red[0] * (1.f / 256.f) + 1e-5f);
        __syncthreads();
        orms[ix] = f2b(o * scale * onorm[t]);
    }
}

// ------------------------------------- host -------------------------------------
extern "C" void kernel_launch(void* const* d_in, const int* in_sizes, int n_in,
                              void* d_out, int out_size, void* d_ws, size_t ws_size,
                              hipStream_t stream) {
    (void)in_sizes; (void)n_in; (void)out_size; (void)ws_size;
    const float* hs    = (const float*)d_in[0];
    const float* Wq    = (const float*)d_in[1];
    const float* Wk    = (const float*)d_in[2];
    const float* Wv    = (const float*)d_in[3];
    const float* Wb    = (const float*)d_in[4];
    const float* cq    = (const float*)d_in[5];
    const float* ck    = (const float*)d_in[6];
    const float* cv    = (const float*)d_in[7];
    const float* w3    = (const float*)d_in[8];
    const float* w15   = (const float*)d_in[9];
    const float* w31   = (const float*)d_in[10];
    const float* km    = (const float*)d_in[11];
    const float* cm    = (const float*)d_in[12];
    const float* w1    = (const float*)d_in[13];
    const float* b1    = (const float*)d_in[14];
    const float* w2    = (const float*)d_in[15];
    const float* b2    = (const float*)d_in[16];
    const float* onorm = (const float*)d_in[17];
    const float* Wo    = (const float*)d_in[18];
    float* out = (float*)d_out;

    const int NTOK = 8192;                           // B*L
    const size_t CHE = (size_t)16 * 64 * 32 * 256;   // 8,388,608 elems (chunked buffers)

    size_t off = 0;
    char* ws = (char*)d_ws;
    auto alloc = [&](size_t bytes) -> char* {
        char* p = ws + off;
        off += (bytes + 255) & ~(size_t)255;
        return p;
    };
    bf16*  hs_bf = (bf16*)alloc((size_t)NTOK * 1024 * 2);
    bf16*  wq_bf = (bf16*)alloc((size_t)1024 * 1024 * 2);
    bf16*  wk_bf = (bf16*)alloc((size_t)1024 * 1024 * 2);
    bf16*  wv_bf = (bf16*)alloc((size_t)1024 * 1024 * 2);
    bf16*  cm_bf = (bf16*)alloc((size_t)1024 * 1024 * 2);
    bf16*  wo_bf = (bf16*)alloc((size_t)1024 * 1024 * 2);
    bf16*  km_bf = (bf16*)alloc((size_t)768 * 256 * 2);
    bf16*  w1p   = (bf16*)alloc((size_t)1056 * 2048 * 2);
    float* raw   = (float*)alloc((size_t)NTOK * 1024 * 4);
    float* q_act = (float*)alloc((size_t)NTOK * 1024 * 4);
    float* k_act = (float*)alloc((size_t)NTOK * 1024 * 4);
    float* v_act = (float*)alloc((size_t)NTOK * 1024 * 4);
    float* beta  = (float*)alloc((size_t)NTOK * 4 * 4);
    // chunked delta buffers (contiguous; sizes all multiples of 256B):
    bf16*  qn_bf  = (bf16*)alloc(CHE * 2);    // \ qn,kn,kb contiguous -> msy alias (3*CHE),
    bf16*  kn_bf  = (bf16*)alloc(CHE * 2);    // |  hdn_bf alias (qn+kn), orms alias (kb)
    bf16*  kb_bf  = (bf16*)alloc(CHE * 2);    // /
    bf16*  kbT_bf = (bf16*)alloc(CHE * 2);    // \ kbT,vbT,kT,u contiguous -> hdn alias (64MB)
    bf16*  vbT_bf = (bf16*)alloc(CHE * 2);    // |  (vbT also reused as msmid_bf)
    bf16*  kT_bf  = (bf16*)alloc(CHE * 2);    // |
    float* u_buf  = (float*)alloc(CHE * 4);   // /
    bf16*  wn_bf  = (bf16*)alloc(CHE * 2);
    bf16*  al_bf  = (bf16*)alloc((size_t)16 * 64 * 1024 * 2);
    float* dlt    = (float*)alloc((size_t)NTOK * 1024 * 4);
    bf16*  gf_bf  = (bf16*)alloc((size_t)NTOK * 1056 * 2);
    // lifetime-based aliases (producers of the aliased region are dead before reuse):
    bf16*  msy_bf   = qn_bf;            // after scan
    float* ms_mid   = q_act;            // after prep
    bf16*  msmid_bf = vbT_bf;           // after km GEMM
    float* ms_out   = k_act;            // after prep
    float* hdn      = (float*)kbT_bf;   // after scan: 8192x2048 fp32 (spans kbT..u)
    bf16*  hdn_bf   = qn_bf;            // after msconv pipeline done (spans qn+kn)
    bf16*  orms_bf  = kb_bf;            // after cm GEMM

    auto cvt = [&](const float* s, bf16* d, size_t n) {
        cvt_f32_bf16_kernel<<<dim3((unsigned)((n + 255) / 256)), dim3(256), 0, stream>>>(s, d, n);
    };
    auto gemm = [&](const bf16* A, const bf16* B, float* C, int M, int N, int K) {
        gemm_bf16_kernel<<<dim3(N / 64, M / 128), dim3(256), 0, stream>>>(A, B, C, M, N, K);
    };

    // 0) precision conversion of activations + weights
    cvt(hs, hs_bf, (size_t)NTOK * 1024);
    cvt(Wq, wq_bf, (size_t)1024 * 1024);
    cvt(Wk, wk_bf, (size_t)1024 * 1024);
    cvt(Wv, wv_bf, (size_t)1024 * 1024);
    cvt(cm, cm_bf, (size_t)1024 * 1024);
    cvt(Wo, wo_bf, (size_t)1024 * 1024);
    cvt(km, km_bf, (size_t)768 * 256);

    // 1) q/k/v projections + causal conv(4) + SiLU
    const size_t NE = (size_t)NTOK * 1024;
    gemm(hs_bf, wq_bf, raw, NTOK, 1024, 1024);
    conv4_silu_kernel<<<dim3((unsigned)(NE / 256)), dim3(256), 0, stream>>>(raw, cq, q_act, NE);
    gemm(hs_bf, wk_bf, raw, NTOK, 1024, 1024);
    conv4_silu_kernel<<<dim3((unsigned)(NE / 256)), dim3(256), 0, stream>>>(raw, ck, k_act, NE);
    gemm(hs_bf, wv_bf, raw, NTOK, 1024, 1024);
    conv4_silu_kernel<<<dim3((unsigned)(NE / 256)), dim3(256), 0, stream>>>(raw, cv, v_act, NE);

    // 2) beta + delta-rule
    beta_kernel<<<dim3(128), dim3(256), 0, stream>>>(hs, Wb, beta, NTOK * 4);
    delta_prep_kernel<<<dim3(NTOK), dim3(128), 0, stream>>>(q_act, k_act, v_act, beta,
                                                            qn_bf, kn_bf, kb_bf,
                                                            kbT_bf, vbT_bf, kT_bf);
    delta_chunk_kernel<<<dim3(1024), dim3(128), 0, stream>>>(qn_bf, kn_bf, kb_bf, kbT_bf, vbT_bf,
                                                             u_buf, wn_bf, al_bf);
    const int SCAN_LDS = 131072 + 65536 + 8192;   // S(f32) + S^T(bf16 shadow) + Ub^T
    (void)hipFuncSetAttribute(reinterpret_cast<const void*>(delta_scan_kernel),
                              hipFuncAttributeMaxDynamicSharedMemorySize, SCAN_LDS);
    delta_scan_kernel<<<dim3(32), dim3(256), SCAN_LDS, stream>>>(qn_bf, wn_bf, al_bf, kT_bf,
                                                                 u_buf, dlt);

    // 3) multiscale conv path
    const size_t NMS = (size_t)NTOK * 3072;
    msconv_kernel<<<dim3((unsigned)(NMS / 256)), dim3(256), 0, stream>>>(v_act, w3, w15, w31,
                                                                         msy_bf, NMS);
    gemm(msy_bf, km_bf, ms_mid, NTOK * 4, 256, 768);        // (b,l,h)x768 @ 768x256
    cvt(ms_mid, msmid_bf, (size_t)NTOK * 1024);
    gemm(msmid_bf, cm_bf, ms_out, NTOK, 1024, 1024);        // channel mixer

    // 4) gating MLP (K padded 1036 -> 1056)
    gate_kernel<<<dim3(NTOK), dim3(256), 0, stream>>>(hs, ms_out, dlt, v_act, gf_bf);
    const size_t NW1 = (size_t)1056 * 2048;
    padw1_kernel<<<dim3((unsigned)((NW1 + 255) / 256)), dim3(256), 0, stream>>>(w1, w1p, NW1);
    gemm(gf_bf, w1p, hdn, NTOK, 2048, 1056);
    const size_t NH2 = (size_t)NTOK * 2048;
    biasgelu_kernel<<<dim3((unsigned)(NH2 / 256)), dim3(256), 0, stream>>>(hdn, b1, hdn_bf, NH2);
    fuse_out_kernel<<<dim3(NTOK), dim3(256), 0, stream>>>(hdn_bf, w2, b2, ms_out, dlt, v_act,
                                                          onorm, orms_bf);

    // 5) output projection
    gemm(orms_bf, wo_bf, out, NTOK, 1024, 1024);
}
